// RNNBaseline_14379550507700
// MI455X (gfx1250) — matrix operand order repaired and checked by reading, compiled, not run on hardware
//
#include <hip/hip_runtime.h>
#include <hip/hip_bf16.h>
#include <stdint.h>

// ---------------------------------------------------------------------------
// MI455X (gfx1250) LSTM LM forward.
//  - GEMMs: bf16 operands, v_wmma_f32_16x16x32_bf16, 128x128x64 tiles,
//    double-buffered LDS fed by GLOBAL_LOAD_ASYNC_TO_LDS_B128 (ASYNCcnt),
//    zero VGPR staging -> no spills.
//  - Recurrent scan: 32 persistent WGs, fp32 W_hh slice resident in LDS
//    (134KB of the 320KB/WGP), h exchanged via L2 + global spin barrier.
// ---------------------------------------------------------------------------

typedef __attribute__((ext_vector_type(16))) __bf16 bf16x16;
typedef __attribute__((ext_vector_type(8)))  __bf16 bf16x8;
typedef __attribute__((ext_vector_type(8)))  float  v8f;

struct BfPair { bf16x8 lo; bf16x8 hi; };

__device__ __forceinline__ unsigned short f32_to_bf16(float f) {
  unsigned u = __float_as_uint(f);
  unsigned r = u + 0x7FFFu + ((u >> 16) & 1u);   // round-to-nearest-even
  return (unsigned short)(r >> 16);
}

// ---------------------------------------------------------------------------
// fp32 -> bf16 bulk convert with zero-padding (n and ntot multiples of 4).
// ---------------------------------------------------------------------------
__global__ void cvt_bf16_pad(const float* __restrict__ src,
                             unsigned short* __restrict__ dst,
                             long n, long ntot) {
  long i4 = ((long)blockIdx.x * blockDim.x + threadIdx.x) * 4;
  if (i4 >= ntot) return;
  uint2 o = make_uint2(0u, 0u);
  if (i4 < n) {
    float4 v = *(const float4*)(src + i4);
    o.x = (unsigned)f32_to_bf16(v.x) | ((unsigned)f32_to_bf16(v.y) << 16);
    o.y = (unsigned)f32_to_bf16(v.z) | ((unsigned)f32_to_bf16(v.w) << 16);
  }
  *(uint2*)(dst + i4) = o;
}

// ---------------------------------------------------------------------------
// Embedding gather straight to bf16: out[s,:] = bf16(embed[x[s],:])
// ---------------------------------------------------------------------------
__global__ void embed_gather_bf16(const int* __restrict__ x,
                                  const float* __restrict__ embed,
                                  unsigned short* __restrict__ out, int D) {
  int s = blockIdx.x;
  int v = x[s];
  for (int d = threadIdx.x; d < D; d += blockDim.x)
    out[(size_t)s * D + d] = f32_to_bf16(embed[(size_t)v * D + d]);
}

__global__ void zero_u32(unsigned* p, int n) {
  int i = blockIdx.x * blockDim.x + threadIdx.x;
  if (i < n) p[i] = 0u;
}

// ---------------------------------------------------------------------------
// C[M,N] = A[M,K] * B[N,K]^T + bias0[N] (+ bias1[N])
// A,B bf16 (B zero-padded to 128-row multiple), C fp32.
// 128x128 tile, K-step 64, 256 threads, double-buffered LDS fed by
// global_load_async_to_lds_b128 (per-lane 16B global->LDS DMA, ASYNCcnt).
// ---------------------------------------------------------------------------
#define TM 128
#define TN 128
#define TK 64
#define LDT 72   // LDS row stride (bf16 elems): 64 + 8 pad -> conflict-free
#define STAGE_ELEMS (TM * LDT + TN * LDT)          // ushorts per buffer
#define STAGE_BYTES (STAGE_ELEMS * 2)              // 36864

// Issue one stage of async global->LDS copies (8 x b128 per thread).
__device__ __forceinline__ void async_stage(const unsigned short* A,
                                            const unsigned short* B,
                                            int mBase, int nBase, int K, int k0,
                                            int t, unsigned ldsA_off, unsigned ldsB_off) {
  #pragma unroll
  for (int it = 0; it < 4; ++it) {
    int slot = t + it * 256;          // 0..1023
    int row  = slot >> 3;             // 0..127
    int c8   = (slot & 7) * 8;        // 0..56 (bf16 cols)
    unsigned loffA = ldsA_off + (unsigned)(row * LDT + c8) * 2u;
    unsigned loffB = ldsB_off + (unsigned)(row * LDT + c8) * 2u;
    unsigned long long ga =
        (unsigned long long)(const void*)(A + (size_t)(mBase + row) * K + k0 + c8);
    unsigned long long gb =
        (unsigned long long)(const void*)(B + (size_t)(nBase + row) * K + k0 + c8);
    asm volatile("global_load_async_to_lds_b128 %0, %1, off"
                 :: "v"(loffA), "v"(ga) : "memory");
    asm volatile("global_load_async_to_lds_b128 %0, %1, off"
                 :: "v"(loffB), "v"(gb) : "memory");
  }
}

__global__ __launch_bounds__(256)
void gemm_bf16_wmma(const unsigned short* __restrict__ A,
                    const unsigned short* __restrict__ B,
                    const float* __restrict__ bias0,
                    const float* __restrict__ bias1,
                    float* __restrict__ C,
                    int M, int N, int K) {
  extern __shared__ unsigned short lds_us[];   // 2 * STAGE_ELEMS ushorts

  const int t    = threadIdx.x;
  const int lane = t & 31;
  const int wave = t >> 5;        // 0..7
  const int wm   = wave >> 2;     // 0..1 : 64-row slab
  const int wn   = wave & 3;      // 0..3 : 32-col slab
  const int mBase = blockIdx.y * TM;
  const int nBase = blockIdx.x * TN;
  const int half  = lane >> 4;    // 0/1
  const int l16   = lane & 15;

  v8f acc[4][2];
  #pragma unroll
  for (int i = 0; i < 4; ++i)
    #pragma unroll
    for (int j = 0; j < 2; ++j)
      #pragma unroll
      for (int e = 0; e < 8; ++e) acc[i][j][e] = 0.f;

  const unsigned ldsB_rel = (unsigned)(TM * LDT) * 2u;   // B region offset in a buffer

  // prologue: stage 0
  async_stage(A, B, mBase, nBase, K, 0, t, 0u, ldsB_rel);
  asm volatile("s_wait_asynccnt 0x0" ::: "memory");
  __syncthreads();

  const int stages = K / TK;
  for (int s = 0; s < stages; ++s) {
    const int cur = s & 1;
    unsigned short* curA = lds_us + cur * STAGE_ELEMS;
    unsigned short* curB = curA + TM * LDT;

    if (s + 1 < stages) {                  // overlap next-stage DMA with WMMAs
      unsigned nb = (unsigned)((cur ^ 1) * STAGE_BYTES);
      async_stage(A, B, mBase, nBase, K, (s + 1) * TK, t, nb, nb + ldsB_rel);
    }

    // ---- 2 WMMA k-iterations (k-step 32) from current buffer ----
    #pragma unroll
    for (int kk = 0; kk < TK; kk += 32) {
      bf16x16 af[4], bfm[2];
      // A frag: lane row M=l16; j<8 -> K=kk+half*8+j, j>=8 -> K=kk+16+half*8+(j-8)
      #pragma unroll
      for (int i = 0; i < 4; ++i) {
        int row = wm * 64 + i * 16 + l16;
        BfPair p;
        p.lo = *(const bf16x8*)(&curA[row * LDT + kk + half * 8]);
        p.hi = *(const bf16x8*)(&curA[row * LDT + kk + 16 + half * 8]);
        af[i] = __builtin_bit_cast(bf16x16, p);
      }
      // B frag: lane col N=l16; j<8 -> K=kk+half*8+j, j>=8 -> K=kk+16+half*8+(j-8)
      #pragma unroll
      for (int j = 0; j < 2; ++j) {
        int col = wn * 32 + j * 16 + l16;
        BfPair p;
        p.lo = *(const bf16x8*)(&curB[col * LDT + kk + half * 8]);
        p.hi = *(const bf16x8*)(&curB[col * LDT + kk + 16 + half * 8]);
        bfm[j] = __builtin_bit_cast(bf16x16, p);
      }
      #pragma unroll
      for (int i = 0; i < 4; ++i)
        #pragma unroll
        for (int j = 0; j < 2; ++j)
          acc[i][j] = __builtin_amdgcn_wmma_f32_16x16x32_bf16(
              false, af[i], false, bfm[j], (short)0, acc[i][j], false, false);
    }

    // next buffer ready before anyone reads it
    asm volatile("s_wait_asynccnt 0x0" ::: "memory");
    __syncthreads();
  }

  // ---- store C (+bias); C/D layout: VGPR r -> row = base + half*8 + r ----
  #pragma unroll
  for (int i = 0; i < 4; ++i) {
    int row0 = mBase + wm * 64 + i * 16 + half * 8;
    #pragma unroll
    for (int j = 0; j < 2; ++j) {
      int col = nBase + wn * 32 + j * 16 + l16;
      if (col < N) {
        float b = bias0 ? bias0[col] : 0.f;
        if (bias1) b += bias1[col];
        #pragma unroll
        for (int r = 0; r < 8; ++r)
          C[(size_t)(row0 + r) * N + col] = acc[i][j][r] + b;
      }
    }
  }
}

// ---------------------------------------------------------------------------
// Persistent multi-WG LSTM scan (fp32 recurrent path, bf16 activation out).
// 32 WGs x 256 threads; each WG owns 4 gates x 16 h-elements (64 rows of
// W_hh resident in LDS, conflict-free padded). Split-K=4 GEMV + shuffles.
// ---------------------------------------------------------------------------
#define NWG  32
#define WSTR 524   // row stride (floats); 524 % 64 = 12
#define QOFF 131   // quarter offset;      131 % 64 = 3 -> banks 3*(4r+q) distinct

__device__ __forceinline__ float sigmf(float x) { return 1.f / (1.f + __expf(-x)); }

__global__ __launch_bounds__(256)
void lstm_scan(const float* __restrict__ pre,          // [S, 2048]
               const float* __restrict__ Whh,          // [2048, 512]
               unsigned short* __restrict__ hout,      // [S, 512] bf16
               float* __restrict__ hbuf,               // [512] exchange
               unsigned* __restrict__ counter,         // monotonic barrier
               int S) {
  extern __shared__ float sm[];
  float* w  = sm;                 // 64*WSTR floats (~134 KB)
  float* h  = sm + 64 * WSTR;     // 512
  float* g4 = h + 512;            // 64
  float* c  = g4 + 64;            // 16

  const int t  = threadIdx.x;
  const int wg = blockIdx.x;

  for (int e = t; e < 64 * 512; e += 256) {
    int rl = e >> 9;
    int k  = e & 511;
    int n  = (rl >> 4) * 512 + wg * 16 + (rl & 15);
    w[rl * WSTR + (k >> 7) * QOFF + (k & 127)] = Whh[(size_t)n * 512 + k];
  }
  for (int i = t; i < 512; i += 256) h[i] = 0.f;
  if (t < 16) c[t] = 0.f;
  __syncthreads();

  unsigned target = NWG;
  const int rl = t >> 2;
  const int kq = t & 3;
  const float* wrow = &w[rl * WSTR + kq * QOFF];
  const float* hq   = &h[kq * 128];

  for (int s = 0; s < S; ++s) {
    float sum = 0.f;
    #pragma unroll 16
    for (int k = 0; k < 128; ++k) sum = __builtin_fmaf(wrow[k], hq[k], sum);
    sum += __shfl_xor(sum, 1, 32);
    sum += __shfl_xor(sum, 2, 32);
    if (kq == 0) g4[rl] = sum;
    __syncthreads();

    if (t < 16) {
      const float* p = pre + (size_t)s * 2048 + wg * 16 + t;
      float iv = sigmf(p[0]    + g4[t]);
      float fv = sigmf(p[512]  + g4[16 + t]);
      float gv = tanhf(p[1024] + g4[32 + t]);
      float ov = sigmf(p[1536] + g4[48 + t]);
      float cc = fv * c[t] + iv * gv;
      c[t] = cc;
      float hv = ov * tanhf(cc);
      hout[(size_t)s * 512 + wg * 16 + t] = f32_to_bf16(hv);
      __hip_atomic_store(&hbuf[wg * 16 + t], hv,
                         __ATOMIC_RELEASE, __HIP_MEMORY_SCOPE_AGENT);
    }
    __syncthreads();

    if (t == 0) {   // barrier A: h(s) published
      __hip_atomic_fetch_add(counter, 1u, __ATOMIC_ACQ_REL, __HIP_MEMORY_SCOPE_AGENT);
      while (__hip_atomic_load(counter, __ATOMIC_ACQUIRE, __HIP_MEMORY_SCOPE_AGENT) < target)
        __builtin_amdgcn_s_sleep(2);
    }
    target += NWG;
    __syncthreads();

    for (int i = t; i < 512; i += 256)
      h[i] = __hip_atomic_load(&hbuf[i], __ATOMIC_RELAXED, __HIP_MEMORY_SCOPE_AGENT);
    __syncthreads();

    if (t == 0) {   // barrier B: h(s) consumed everywhere
      __hip_atomic_fetch_add(counter, 1u, __ATOMIC_ACQ_REL, __HIP_MEMORY_SCOPE_AGENT);
      while (__hip_atomic_load(counter, __ATOMIC_ACQUIRE, __HIP_MEMORY_SCOPE_AGENT) < target)
        __builtin_amdgcn_s_sleep(2);
    }
    target += NWG;
    __syncthreads();
  }
}

// ---------------------------------------------------------------------------
extern "C" void kernel_launch(void* const* d_in, const int* in_sizes, int n_in,
                              void* d_out, int out_size, void* d_ws, size_t ws_size,
                              hipStream_t stream) {
  const int S = 4096, D = 512, H = 512, L = 4, V = 50257;
  const int G = 4 * H;                 // 2048
  const int Vpad = 50304;              // 393 * 128

  const int*   x      = (const int*)d_in[0];
  const float* embed  = (const float*)d_in[1];
  const float* W_ih   = (const float*)d_in[2];
  const float* W_hh   = (const float*)d_in[3];
  const float* b_ih   = (const float*)d_in[4];
  const float* b_hh   = (const float*)d_in[5];
  const float* head_w = (const float*)d_in[6];
  const float* head_b = (const float*)d_in[7];
  float* out = (float*)d_out;

  // workspace layout (bytes)
  char* w8 = (char*)d_ws;
  unsigned short* actA   = (unsigned short*)w8;                    w8 += (size_t)S * D * 2;
  unsigned short* actB   = (unsigned short*)w8;                    w8 += (size_t)S * D * 2;
  float*          pre    = (float*)w8;                             w8 += (size_t)S * G * 4;
  unsigned short* wihb   = (unsigned short*)w8;                    w8 += (size_t)G * D * 2;
  unsigned short* headwb = (unsigned short*)w8;                    w8 += (size_t)Vpad * D * 2;
  float*          hbuf   = (float*)w8;                             w8 += 512 * 4;
  unsigned*       counters = (unsigned*)w8;

  zero_u32<<<1, 32, 0, stream>>>(counters, 8);
  embed_gather_bf16<<<S, 256, 0, stream>>>(x, embed, actA, D);

  { // head weights -> bf16, zero-padded to Vpad rows
    long n = (long)V * D, ntot = (long)Vpad * D;
    cvt_bf16_pad<<<(unsigned)((ntot / 4 + 255) / 256), 256, 0, stream>>>(head_w, headwb, n, ntot);
  }

  const size_t scan_lds = (size_t)(64 * WSTR + 512 + 64 + 16) * sizeof(float);
  const size_t gemm_lds = 2 * STAGE_BYTES;   // 73728 B (double buffer)

  unsigned short* hin  = actA;
  unsigned short* hnxt = actB;
  for (int l = 0; l < L; ++l) {
    long n = (long)G * D;
    cvt_bf16_pad<<<(unsigned)((n / 4 + 255) / 256), 256, 0, stream>>>(
        W_ih + (size_t)l * G * D, wihb, n, n);
    dim3 gg(G / TN, S / TM);                 // (16, 32)
    gemm_bf16_wmma<<<gg, 256, gemm_lds, stream>>>(
        hin, wihb, b_ih + (size_t)l * G, b_hh + (size_t)l * G, pre, S, G, D);
    lstm_scan<<<NWG, 256, scan_lds, stream>>>(
        pre, W_hh + (size_t)l * G * H, hnxt, hbuf, counters + l, S);
    unsigned short* tmp = hin; hin = hnxt; hnxt = tmp;
  }

  dim3 gh(Vpad / TN, S / TM);                // (393, 32)
  gemm_bf16_wmma<<<gh, 256, gemm_lds, stream>>>(
      hin, headwb, head_b, nullptr, out, S, V, D);
}
